// DLASSO_unfolded_10677288698530
// MI455X (gfx1250) — matrix operand order repaired and checked by reading, not compiled, not run on hardware
//
#include <hip/hip_runtime.h>
#include <math.h>

typedef __attribute__((ext_vector_type(2)))  float    v2f;
typedef __attribute__((ext_vector_type(8)))  float    v8f;
typedef __attribute__((ext_vector_type(8)))  _Float16 v8h;
typedef __attribute__((ext_vector_type(16))) _Float16 v16h;

#define P_    256
#define Bb    8
#define MD    256
#define ND    512
#define KIT   15
#define EDGES 2048                 // directed edges per batch (2*E_UND)
#define BPN   (Bb * P_ * ND)       // 1048576 elements per (y/U/delta) buffer

// ---------------------------------------------------------------------------
// Stage 1: AtA[p] = A0[p]^T (512x256) * A0[p] (256x512), f32 WMMA, store f16.
// One wave computes a 64x64 output block = 4x4 tiles of V_WMMA_F32_16X16X4_F32,
// so each k-step amortizes 8 operand fetches over 16 WMMAs (1 b32 load/wmma).
// ---------------------------------------------------------------------------
__global__ __launch_bounds__(256) void k_ata(const float* __restrict__ A,
                                             _Float16* __restrict__ AtAh) {
  const int lane = threadIdx.x & 31;
  const int wid  = blockIdx.x * (blockDim.x >> 5) + (threadIdx.x >> 5);
  const int p    = wid >> 6;            // 8x8 = 64 superblocks of 64x64 per p
  const int sb   = wid & 63;
  const int bi   = sb >> 3, bj = sb & 7;
  const int nbase = bi * 64, lbase = bj * 64;
  const float* Ap = A + (size_t)p * MD * ND;
  const int hf   = lane >> 4;           // lane half selects K-pair (0,1)/(2,3)
  const int l15  = lane & 15;
  v8f acc[4][4] = {};
  for (int kk = 0; kk < 64; ++kk) {
    const int m = kk * 4 + hf * 2;      // contraction index (rows of A0)
    const float* r0 = Ap + (size_t)m * ND;
    const float* r1 = r0 + ND;
    v2f a[4], b[4];
    #pragma unroll
    for (int i = 0; i < 4; ++i) {
      const int nc = nbase + i * 16 + l15;
      const int lc = lbase + i * 16 + l15;
      a[i].x = r0[nc]; a[i].y = r1[nc]; // A_op[M,K] = A0[m, n]
      b[i].x = r0[lc]; b[i].y = r1[lc]; // B_op[K,N] = A0[m, l]
    }
    #pragma unroll
    for (int i = 0; i < 4; ++i)
      #pragma unroll
      for (int j = 0; j < 4; ++j)
        acc[i][j] = __builtin_amdgcn_wmma_f32_16x16x4_f32(
            false, a[i], false, b[j], (short)0, acc[i][j], false, false);
  }
  _Float16* out = AtAh + (size_t)p * ND * ND;
  #pragma unroll
  for (int i = 0; i < 4; ++i)
    #pragma unroll
    for (int r = 0; r < 8; ++r) {
      const int n = nbase + i * 16 + r + hf * 8;  // D layout: M = r (+8 hi half)
      #pragma unroll
      for (int j = 0; j < 4; ++j)
        out[(size_t)n * ND + lbase + j * 16 + l15] = (_Float16)acc[i][j][r];
    }
}

// ---------------------------------------------------------------------------
// Atb[b,p,n] = sum_m A0[p,m,n] * b[b,p,m]   (coalesced over n)
// ---------------------------------------------------------------------------
__global__ void k_atb(const float* __restrict__ A, const float* __restrict__ bv,
                      float* __restrict__ Atb) {
  const int idx = blockIdx.x * blockDim.x + threadIdx.x;
  if (idx >= BPN) return;
  const int n = idx & (ND - 1);
  const int p = (idx >> 9) & (P_ - 1);
  const int b = idx >> 17;
  const float* Ap = A + (size_t)p * MD * ND + n;
  const float* bp = bv + ((size_t)b * P_ + p) * MD;
  float s = 0.f;
  for (int m = 0; m < MD; ++m) s += Ap[(size_t)m * ND] * bp[m];
  Atb[idx] = s;
}

// ---------------------------------------------------------------------------
// degrees, hyper-parameter table (cumsum+sigmoid), hyp_last output
// ---------------------------------------------------------------------------
__global__ void k_misc(const int* __restrict__ edges, const float* __restrict__ param,
                       const float* __restrict__ maxp, float* __restrict__ deg,
                       float* __restrict__ hyp, float* __restrict__ out_hyp_last) {
  const int tid = blockIdx.x * blockDim.x + threadIdx.x;
  if (tid < Bb * P_) {
    const int b = tid >> 8, p = tid & 255;
    const int* src = edges + (size_t)b * 2 * EDGES;   // edge_index[b,0,:]
    int c = 0;
    for (int e = 0; e < EDGES; ++e) c += (src[e] == p);
    deg[tid] = (float)c;
  }
  if (tid < P_ * 4) {
    float cum = 0.f;
    const float mp = maxp[tid & 3];
    for (int k = 0; k < KIT; ++k) {
      cum += param[k * P_ * 4 + tid];
      float h = mp / (1.f + expf(-cum));
      h = fminf(fmaxf(h, 1e-4f), 0.99f);
      hyp[k * P_ * 4 + tid] = h;
      if (k == KIT - 1) out_hyp_last[tid] = h;
    }
  }
}

__global__ void k_initvec(const float* __restrict__ y0, float* __restrict__ yf,
                          _Float16* __restrict__ yh) {
  const int i = blockIdx.x * blockDim.x + threadIdx.x;
  if (i < BPN) { const float v = y0[i]; yf[i] = v; yh[i] = (_Float16)v; }
}

// ---------------------------------------------------------------------------
// Per-iteration fused step: AtAy via V_WMMA_F32_16X16X32_F16 (D cols = batch),
// U update (folded from previous step), grad/clip/y update, ping-pong f16 y,
// zero delta accumulator, write Y[k] slice.
// One wave per (p, 16-row n-tile). 16 k-steps of K=32. L2-bandwidth bound on
// the 128 MB f16 AtA (fits the 192 MB global L2).
// ---------------------------------------------------------------------------
__global__ __launch_bounds__(256) void k_step(
    const _Float16* __restrict__ AtAh, const float* __restrict__ Atb,
    const float* __restrict__ deg, const float* __restrict__ hypk,
    const float* __restrict__ hypprev,
    float* __restrict__ yf, const _Float16* __restrict__ yh_src,
    _Float16* __restrict__ yh_dst,
    const float* __restrict__ U_in, float* __restrict__ U_out,
    const float* __restrict__ delta_in, float* __restrict__ delta_ws,
    float* __restrict__ Yout, int k, float gm, float ym, float ym_prev) {
  const int lane = threadIdx.x & 31;
  const int wid  = blockIdx.x * (blockDim.x >> 5) + (threadIdx.x >> 5);
  const int p    = wid >> 5;            // 32 n-tiles per p
  const int nt   = wid & 31;
  const int hf   = lane >> 4;
  const int l15  = lane & 15;
  const int n0   = nt * 16 + l15;       // A-operand row (M = l15 for both halves)
  const _Float16* Arow = AtAh + ((size_t)p * ND + n0) * ND;
  const int koffA = hf * 8;             // K split 0..7/16..23 vs 8..15/24..31
  const int koffB = hf * 8;             // B: K split 0..7/16..23 vs 8..15/24..31
  const int bcol  = l15;                // D/B column = batch index (8..15 pad)
  const _Float16* yrow = yh_src + ((size_t)(bcol & 7) * P_ + p) * ND;
  v8f acc = {};
  for (int ks = 0; ks < ND / 32; ++ks) {
    const int lb = ks * 32;
    v16h a, bvv;
    const v8h alo = *(const v8h*)(Arow + lb + koffA);
    const v8h ahi = *(const v8h*)(Arow + lb + 16 + koffA);
    #pragma unroll
    for (int i = 0; i < 8; ++i) { a[i] = alo[i]; a[i + 8] = ahi[i]; }
    if (bcol < 8) {
      const v8h blo = *(const v8h*)(yrow + lb + koffB);
      const v8h bhi = *(const v8h*)(yrow + lb + 16 + koffB);
      #pragma unroll
      for (int i = 0; i < 8; ++i) { bvv[i] = blo[i]; bvv[i + 8] = bhi[i]; }
    } else {
      #pragma unroll
      for (int i = 0; i < 16; ++i) bvv[i] = (_Float16)0.f;
    }
    acc = __builtin_amdgcn_wmma_f32_16x16x32_f16(false, a, false, bvv,
                                                 (short)0, acc, false, false);
  }
  if (bcol >= 8) return;                // padded columns: no epilogue work
  const int b = bcol;
  const float alpha = hypk[p * 4 + 0], tau = hypk[p * 4 + 1], rho = hypk[p * 4 + 2];
  const float eta_p = hypprev[p * 4 + 3];
  const float dg    = deg[b * P_ + p];
  #pragma unroll
  for (int r = 0; r < 8; ++r) {
    const int n = nt * 16 + r + hf * 8; // D layout: M = r (+8 for hi lanes)
    const size_t idx = ((size_t)b * P_ + p) * ND + n;
    const float atb = Atb[idx];
    const float yv  = yf[idx];
    const float del = delta_in[idx];
    const float Uv  = U_in[idx];
    float uc = Uv;
    if (k > 0) {                        // fold previous step's U update
      uc = Uv + del * eta_p;
      uc = fminf(fmaxf(uc, -ym_prev), ym_prev);
    }
    const float sg = (yv > 0.f) ? 1.f : ((yv < 0.f) ? -1.f : 0.f);
    float g = acc[r] - atb + sg * tau + uc * dg + del * rho;
    g = fminf(fmaxf(g, -gm), gm);
    float yn = yv - alpha * g;
    yn = fminf(fmaxf(yn, -ym), ym);
    yf[idx]     = yn;
    yh_dst[idx] = (_Float16)yn;
    U_out[idx]  = uc;
    delta_ws[idx] = 0.f;                // accumulator for this step's scatter
    Yout[idx]   = yn;
  }
}

// ---------------------------------------------------------------------------
// Graph scatter: delta[s] += y[s]-y[d]; delta[d] -= y[s]-y[d]
// One block per directed edge, 512-vector split over threads.
// ---------------------------------------------------------------------------
__global__ void k_edge(const int* __restrict__ edges, const float* __restrict__ yf,
                       float* __restrict__ delta) {
  const int eb = blockIdx.x;            // b*EDGES + e
  const int b = eb >> 11, e = eb & (EDGES - 1);
  const int s = edges[(size_t)b * 2 * EDGES + e];
  const int d = edges[(size_t)b * 2 * EDGES + EDGES + e];
  const float* ys = yf + ((size_t)b * P_ + s) * ND;
  const float* yd = yf + ((size_t)b * P_ + d) * ND;
  float* ds = delta + ((size_t)b * P_ + s) * ND;
  float* dd = delta + ((size_t)b * P_ + d) * ND;
  for (int n = threadIdx.x; n < ND; n += blockDim.x) {
    const float diff = ys[n] - yd[n];
    atomicAdd(ds + n, diff);
    atomicAdd(dd + n, -diff);
  }
}

extern "C" void kernel_launch(void* const* d_in, const int* in_sizes, int n_in,
                              void* d_out, int out_size, void* d_ws, size_t ws_size,
                              hipStream_t stream) {
  const float* A      = (const float*)d_in[0];
  const float* bvec   = (const float*)d_in[1];
  const float* param  = (const float*)d_in[2];
  const float* maxp   = (const float*)d_in[3];
  const float* y0     = (const float*)d_in[4];
  const float* U0     = (const float*)d_in[5];
  const float* delta0 = (const float*)d_in[6];
  const int*   edges  = (const int*)d_in[7];
  float* out = (float*)d_out;

  char* ws = (char*)d_ws;
  _Float16* AtAh = (_Float16*)ws;  ws += (size_t)P_ * ND * ND * 2;  // 128 MB, L2-resident
  float*    Atb  = (float*)ws;     ws += (size_t)BPN * 4;
  float*    yf   = (float*)ws;     ws += (size_t)BPN * 4;
  _Float16* yh0  = (_Float16*)ws;  ws += (size_t)BPN * 2;
  _Float16* yh1  = (_Float16*)ws;  ws += (size_t)BPN * 2;
  float*    Uw   = (float*)ws;     ws += (size_t)BPN * 4;
  float*    dw   = (float*)ws;     ws += (size_t)BPN * 4;
  float*    deg  = (float*)ws;     ws += (size_t)Bb * P_ * 4;
  float*    hyp  = (float*)ws;

  k_ata<<<2048, 256, 0, stream>>>(A, AtAh);   // 256 p * 64 blocks = 16384 waves
  k_atb<<<BPN / 256, 256, 0, stream>>>(A, bvec, Atb);
  k_misc<<<8, 256, 0, stream>>>(edges, param, maxp, deg, hyp, out + (size_t)KIT * BPN);
  k_initvec<<<BPN / 256, 256, 0, stream>>>(y0, yf, yh0);

  _Float16* ysrc = yh0; _Float16* ydst = yh1;
  for (int k = 0; k < KIT; ++k) {
    const float gm  = fmaxf(1.f, 30.f - (float)k);
    const float ym  = fmaxf(10.f, 200.f - 3.f * (float)k);
    const float ymp = fmaxf(10.f, 200.f - 3.f * (float)(k - 1));
    const float* hk  = hyp + k * P_ * 4;
    const float* hp  = hyp + (k > 0 ? k - 1 : 0) * P_ * 4;
    const float* Uin = (k == 0) ? U0 : Uw;
    const float* din = (k == 0) ? delta0 : dw;
    k_step<<<1024, 256, 0, stream>>>(AtAh, Atb, deg, hk, hp, yf, ysrc, ydst,
                                     Uin, Uw, din, dw, out + (size_t)k * BPN,
                                     k, gm, ym, ymp);
    k_edge<<<Bb * EDGES, 256, 0, stream>>>(edges, yf, dw);
    _Float16* t = ysrc; ysrc = ydst; ydst = t;
  }
}
